// CTAM_62706522521792
// MI455X (gfx1250) — matrix operand order repaired and checked by reading, not compiled, not run on hardware
//
#include <hip/hip_runtime.h>
#include <hip/hip_bf16.h>
#include <math.h>

// ---------------------------------------------------------------------------
// CDNA5 (gfx1250): bf16 WMMA everywhere, bf16 intermediates, async global->LDS
// staging (GLOBAL_LOAD_ASYNC_TO_LDS_B128 + s_wait_asynccnt).
// ---------------------------------------------------------------------------

typedef __attribute__((ext_vector_type(16))) __bf16 v16bf;
typedef __attribute__((ext_vector_type(8)))  __bf16 v8bf;
typedef __attribute__((ext_vector_type(8)))  float  v8f;

__device__ __forceinline__ v8f wmma_bf16(v16bf a, v16bf b, v8f c) {
  return __builtin_amdgcn_wmma_f32_16x16x32_bf16(false, a, false, b,
                                                 (short)0, c, false, false);
}

// A fragment (16x32, MxK): lane holds K=hi*16..+15
__device__ __forceinline__ v16bf frag_a_lds(const __bf16* row, int hi) {
  v16bf f;
  *((v8bf*)&f)     = *(const v8bf*)(row + hi * 16);
  *((v8bf*)&f + 1) = *(const v8bf*)(row + hi * 16 + 8);
  return f;
}
// B fragment (32x16, KxN) from [n][k] storage: lane n=lane%16 holds K=hi*16..+15
__device__ __forceinline__ v16bf frag_b_lds(const __bf16* nrow, int hi) {
  v16bf f;
  *((v8bf*)&f)     = *(const v8bf*)(nrow + hi * 16);
  *((v8bf*)&f + 1) = *(const v8bf*)(nrow + hi * 16 + 8);
  return f;
}
__device__ __forceinline__ v16bf frag_from_g(const __bf16* p0, const __bf16* p1) {
  v16bf f;
  *((v8bf*)&f)     = *(const v8bf*)p0;
  *((v8bf*)&f + 1) = *(const v8bf*)p1;
  return f;
}

// Async global->LDS 16B copy (per lane). Generic LDS pointer's low 32 bits are
// the LDS offset (aperture in high bits), which is what VDST wants.
__device__ __forceinline__ unsigned lds_off(const void* p) {
  return (unsigned)(unsigned long long)p;
}
__device__ __forceinline__ void async_ld_b128(const void* gptr, unsigned loff) {
  asm volatile("global_load_async_to_lds_b128 %0, %1, off"
               :: "v"(loff), "v"(gptr) : "memory");
}
__device__ __forceinline__ void wait_async() {
  asm volatile("s_wait_asynccnt 0x0" ::: "memory");
}

// ---------------------------------------------------------------------------
// bf16 GEMM: C[b] = (A[b] @ W^T + bias) * oscale
//   A bf16 [M][K] per batch, W bf16 [N][K].
//   EPI 0: bf16 store C[b*sC + m*N + n]
//   EPI 1: bf16 transposed store C[b*sC + n*M + m] (packed 16B along m)
//   EPI 2: fp32 transposed store + BN + SiLU + residual (final output)
// Block 256 thr (8 waves). Tile 128(M) x 64(N), K-chunks of 32. Async staging.
// ---------------------------------------------------------------------------
template<int EPI>
__global__ __launch_bounds__(256)
void gemm_bf16(const __bf16* __restrict__ A, long sA,
               const __bf16* __restrict__ W,
               const float* __restrict__ bias, float oscale,
               void* __restrict__ Cv, long sC,
               int M, int N, int K,
               const float* __restrict__ tgt,
               const float* __restrict__ bng, const float* __restrict__ bnb,
               const float* __restrict__ bnm, const float* __restrict__ bnv) {
  __shared__ __attribute__((aligned(16))) __bf16 As[128][32];
  __shared__ __attribute__((aligned(16))) __bf16 Bs[64][32];

  const int b = blockIdx.z;
  const int m0 = blockIdx.y * 128, n0 = blockIdx.x * 64;
  const int tid = threadIdx.x, lane = tid & 31, wv = tid >> 5;
  const int ml = lane & 15, hi = lane >> 4;
  const int mq = wv & 3, nq = wv >> 2;      // wave tile: rows mq*32..+31, cols nq*32..+31
  const __bf16* Ab = A + (long)b * sA;
  v8f acc[2][2] = {};

  for (int kc = 0; kc < K; kc += 32) {
    __syncthreads();
    // A chunk: 128x32 bf16 = 512 x 8-elem pieces, 2 per thread, async
#pragma unroll
    for (int u = 0; u < 2; ++u) {
      const int ch = tid * 2 + u;
      const int r = ch >> 2, c8 = (ch & 3) * 8;
      async_ld_b128(Ab + (long)(m0 + r) * K + kc + c8, lds_off(&As[r][c8]));
    }
    // B chunk: 64x32 bf16 = 256 pieces, 1 per thread, async
    {
      const int r = tid >> 2, c8 = (tid & 3) * 8;
      async_ld_b128(W + (long)(n0 + r) * K + kc + c8, lds_off(&Bs[r][c8]));
    }
    wait_async();
    __syncthreads();

    v16bf af0 = frag_a_lds(&As[mq * 32 + ml][0], hi);
    v16bf af1 = frag_a_lds(&As[mq * 32 + 16 + ml][0], hi);
    v16bf bf0 = frag_b_lds(&Bs[nq * 32 + ml][0], hi);
    v16bf bf1 = frag_b_lds(&Bs[nq * 32 + 16 + ml][0], hi);
    acc[0][0] = wmma_bf16(af0, bf0, acc[0][0]);
    acc[0][1] = wmma_bf16(af0, bf1, acc[0][1]);
    acc[1][0] = wmma_bf16(af1, bf0, acc[1][0]);
    acc[1][1] = wmma_bf16(af1, bf1, acc[1][1]);
  }

#pragma unroll
  for (int i = 0; i < 2; ++i)
#pragma unroll
    for (int j = 0; j < 2; ++j) {
      const int n = n0 + nq * 32 + j * 16 + ml;
      const int mb = m0 + mq * 32 + i * 16 + hi * 8;  // D layout: M = r + hi*8
      const float bv = bias ? bias[n] : 0.0f;
      if constexpr (EPI == 0) {
        __bf16* C = (__bf16*)Cv;
#pragma unroll
        for (int r = 0; r < 8; ++r)
          C[(long)b * sC + (long)(mb + r) * N + n] =
              (__bf16)((acc[i][j][r] + bv) * oscale);
      } else if constexpr (EPI == 1) {
        __bf16* C = (__bf16*)Cv;
        v8bf pk;
#pragma unroll
        for (int r = 0; r < 8; ++r) pk[r] = (__bf16)((acc[i][j][r] + bv) * oscale);
        *(v8bf*)&C[(long)b * sC + (long)n * M + mb] = pk;   // m contiguous
      } else {
        float* C = (float*)Cv;
        const float inv = bng[n] * rsqrtf(bnv[n] + 1e-5f);
#pragma unroll
        for (int r = 0; r < 8; ++r) {
          const long idx = (long)b * sC + (long)n * M + mb + r;
          const float bn = (acc[i][j][r] - bnm[n]) * inv + bnb[n];
          const float sig = 1.0f / (1.0f + __expf(-bn));
          C[idx] = tgt[idx] + bn * sig;
        }
      }
    }
}

// ---------------------------------------------------------------------------
// Attention: block = (b, h, 16-query strip). All operands bf16, zero cvt on
// the WMMA input path. S (16x1024) in LDS bf16, exact in-place softmax,
// exact per-key column sums for attn_map.
// ---------------------------------------------------------------------------
__global__ __launch_bounds__(256)
void attn_kernel(const __bf16* __restrict__ qp, const __bf16* __restrict__ kp,
                 const __bf16* __restrict__ vt, __bf16* __restrict__ ctx,
                 float* __restrict__ colsum) {
  constexpr int L = 1024, E = 256, NH = 8;
  const int qb = blockIdx.x, h = blockIdx.y, b = blockIdx.z;
  const int tid = threadIdx.x, lane = tid & 31, wv = tid >> 5;
  const int ml = lane & 15, hi = lane >> 4;

  __shared__ __attribute__((aligned(16))) __bf16 S[16][L];   // 32 KB
  __shared__ float red[16][17];
  __shared__ float ctxs[16][32];

  // Q fragment (1/sqrt(32) already folded into qp by its GEMM epilogue)
  const __bf16* qrow = qp + ((long)b * L + qb * 16 + ml) * E + h * 32;
  v16bf qf = frag_from_g(qrow + hi * 16, qrow + hi * 16 + 8);

  // scores: wave w owns key tiles kt = w + 8t
#pragma unroll
  for (int t = 0; t < 8; ++t) {
    const int kt = wv + 8 * t;
    const __bf16* krow = kp + ((long)b * L + kt * 16 + ml) * E + h * 32 + hi * 16;
    v16bf kf = frag_from_g(krow, krow + 8);
    v8f s = {};
    s = wmma_bf16(qf, kf, s);
#pragma unroll
    for (int r = 0; r < 8; ++r) S[hi * 8 + r][kt * 16 + ml] = (__bf16)s[r];
  }
  __syncthreads();

  // exact softmax over 1024 keys; 16 threads per row
  const int row = tid >> 4, seg = tid & 15;
  float mx = -1e30f;
  for (int c = seg * 64; c < seg * 64 + 64; ++c)
    mx = fmaxf(mx, (float)S[row][c]);
  red[row][seg] = mx;
  __syncthreads();
  if (seg == 0) {
    float m2 = -1e30f;
    for (int i = 0; i < 16; ++i) m2 = fmaxf(m2, red[row][i]);
    red[row][16] = m2;
  }
  __syncthreads();
  mx = red[row][16];
  float sum = 0.0f;
  for (int c = seg * 64; c < seg * 64 + 64; ++c)
    sum += __expf((float)S[row][c] - mx);
  __syncthreads();
  red[row][seg] = sum;
  __syncthreads();
  if (seg == 0) {
    float s2 = 0.0f;
    for (int i = 0; i < 16; ++i) s2 += red[row][i];
    red[row][16] = s2;
  }
  __syncthreads();
  const float invs = 1.0f / red[row][16];
  for (int c = seg * 64; c < seg * 64 + 64; ++c)
    S[row][c] = (__bf16)(__expf((float)S[row][c] - mx) * invs);
  __syncthreads();

  // exact column sums (attn_map numerator)
#pragma unroll
  for (int cc = 0; cc < 4; ++cc) {
    const int c = tid * 4 + cc;
    float cs = 0.0f;
    for (int r = 0; r < 16; ++r) cs += (float)S[r][c];
    atomicAdd(&colsum[(long)b * L + c], cs);
  }

  // ctx = P @ V ; wave w owns 32-key chunks {w, w+8, w+16, w+24}
  for (int i = tid; i < 16 * 32; i += 256) ((float*)ctxs)[i] = 0.0f;
  __syncthreads();

  v8f a0 = {}, a1 = {};
#pragma unroll
  for (int t = 0; t < 4; ++t) {
    const int kc = (wv + 8 * t) * 32;
    v16bf pf = frag_from_g(&S[ml][kc + hi * 16], &S[ml][kc + hi * 16 + 8]);
#pragma unroll
    for (int dt = 0; dt < 2; ++dt) {
      const __bf16* vrow =
          vt + ((long)(b * NH + h) * 32 + dt * 16 + ml) * L + kc + hi * 16;
      v16bf vf = frag_from_g(vrow, vrow + 8);
      if (dt == 0) a0 = wmma_bf16(pf, vf, a0);
      else         a1 = wmma_bf16(pf, vf, a1);
    }
  }
#pragma unroll
  for (int r = 0; r < 8; ++r) {
    atomicAdd(&ctxs[hi * 8 + r][ml], a0[r]);
    atomicAdd(&ctxs[hi * 8 + r][16 + ml], a1[r]);
  }
  __syncthreads();

  for (int i = tid; i < 512; i += 256) {
    const int r = i >> 5, d = i & 31;
    ctx[((long)b * L + qb * 16 + r) * E + h * 32 + d] = (__bf16)ctxs[r][d];
  }
}

// ---------------------------------------------------------------------------
__global__ void cvt_kernel(const float* __restrict__ in, __bf16* __restrict__ out,
                           int n) {
  int i = blockIdx.x * 256 + threadIdx.x;
  if (i < n) out[i] = (__bf16)in[i];
}

// fp32 [b][C][HW] -> bf16 [b][HW][C]
__global__ void transpose_cvt_kernel(const float* __restrict__ in,
                                     __bf16* __restrict__ out, int C, int HW) {
  __shared__ float t[32][33];
  const int b = blockIdx.z;
  const float* ib = in + (long)b * C * HW;
  __bf16* ob = out + (long)b * C * HW;
  const int hw0 = blockIdx.x * 32, c0 = blockIdx.y * 32;
  const int tx = threadIdx.x, ty = threadIdx.y;
  for (int i = ty; i < 32; i += 8)
    t[i][tx] = ib[(long)(c0 + i) * HW + hw0 + tx];
  __syncthreads();
  for (int i = ty; i < 32; i += 8)
    ob[(long)(hw0 + i) * C + c0 + tx] = (__bf16)t[tx][i];
}

__global__ void zero_kernel(float* p, int n) {
  int i = blockIdx.x * blockDim.x + threadIdx.x;
  if (i < n) p[i] = 0.0f;
}
__global__ void attnmap_kernel(const float* __restrict__ colsum,
                               float* __restrict__ out, int n) {
  int i = blockIdx.x * blockDim.x + threadIdx.x;
  if (i < n) out[i] = colsum[i] * (1.0f / 8192.0f);  // mean over 8 heads * 1024 q
}

// ---------------------------------------------------------------------------
extern "C" void kernel_launch(void* const* d_in, const int* in_sizes, int n_in,
                              void* d_out, int out_size, void* d_ws, size_t ws_size,
                              hipStream_t stream) {
  (void)in_sizes; (void)n_in; (void)out_size; (void)ws_size;
  const float* tgt    = (const float*)d_in[0];
  const float* src    = (const float*)d_in[1];
  const float* kv_w   = (const float*)d_in[2];
  const float* ipw    = (const float*)d_in[3];
  const float* ipb    = (const float*)d_in[4];
  const float* opw    = (const float*)d_in[5];
  const float* opb    = (const float*)d_in[6];
  const float* fuse_w = (const float*)d_in[7];
  const float* bng    = (const float*)d_in[8];
  const float* bnb    = (const float*)d_in[9];
  const float* bnm    = (const float*)d_in[10];
  const float* bnvr   = (const float*)d_in[11];
  float* out = (float*)d_out;

  char* wsb = (char*)d_ws;
  auto carve = [&](size_t bytes) {
    char* p = wsb;
    wsb += (bytes + 255) & ~(size_t)255;
    return p;
  };
  const long SEQ = 8L * 1024 * 256;                 // elements
  __bf16* qseq   = (__bf16*)carve(SEQ * 2);         // [b][hw][256] from tgt
  __bf16* sseq   = (__bf16*)carve(8L * 1024 * 128 * 2);  // [b][hw][128] from src
  __bf16* kseq   = (__bf16*)carve(SEQ * 2);
  __bf16* qp     = (__bf16*)carve(SEQ * 2);
  __bf16* kp     = (__bf16*)carve(SEQ * 2);
  __bf16* vt     = (__bf16*)carve(SEQ * 2);         // [(b*256+e)][1024]
  __bf16* ctx    = (__bf16*)carve(SEQ * 2);
  __bf16* outsq  = (__bf16*)carve(SEQ * 2);
  __bf16* w_ip   = (__bf16*)carve(768L * 256 * 2);
  __bf16* w_kv   = (__bf16*)carve(256L * 128 * 2);
  __bf16* w_op   = (__bf16*)carve(256L * 256 * 2);
  __bf16* w_fu   = (__bf16*)carve(256L * 256 * 2);
  float*  colsum = (float*)carve(8192 * 4);

  dim3 blk(256);
  dim3 gg(4, 8, 8);          // N/64, M/128, B  (M=1024, N=256)
  const long sSeq = 1024L * 256;
  const float qscale = 0.17677669529663688f;   // 1/sqrt(32)
  const float* nil = nullptr;

  // weight + activation conversions (bf16, transposed activations)
  cvt_kernel<<<dim3((768 * 256 + 255) / 256), blk, 0, stream>>>(ipw, w_ip, 768 * 256);
  cvt_kernel<<<dim3(128), blk, 0, stream>>>(kv_w, w_kv, 256 * 128);
  cvt_kernel<<<dim3(256), blk, 0, stream>>>(opw, w_op, 256 * 256);
  cvt_kernel<<<dim3(256), blk, 0, stream>>>(fuse_w, w_fu, 256 * 256);
  transpose_cvt_kernel<<<dim3(32, 8, 8), dim3(32, 8), 0, stream>>>(tgt, qseq, 256, 1024);
  transpose_cvt_kernel<<<dim3(32, 4, 8), dim3(32, 8), 0, stream>>>(src, sseq, 128, 1024);
  zero_kernel<<<dim3(32), blk, 0, stream>>>(colsum, 8192);

  // kseq = src_seq @ kv_w^T
  gemm_bf16<0><<<gg, blk, 0, stream>>>(sseq, 1024L * 128, w_kv, nil, 1.0f,
                                       kseq, sSeq, 1024, 256, 128,
                                       nil, nil, nil, nil, nil);
  // qp = (q_seq @ Wq^T + bq) * 1/sqrt(hd)
  gemm_bf16<0><<<gg, blk, 0, stream>>>(qseq, sSeq, w_ip, ipb, qscale,
                                       qp, sSeq, 1024, 256, 256,
                                       nil, nil, nil, nil, nil);
  // kp = kseq @ Wk^T + bk
  gemm_bf16<0><<<gg, blk, 0, stream>>>(kseq, sSeq, w_ip + 256L * 256, ipb + 256,
                                       1.0f, kp, sSeq, 1024, 256, 256,
                                       nil, nil, nil, nil, nil);
  // vt = (kseq @ Wv^T + bv)^T   (per-head V transposed: [e][k])
  gemm_bf16<1><<<gg, blk, 0, stream>>>(kseq, sSeq, w_ip + 2L * 256 * 256, ipb + 512,
                                       1.0f, vt, sSeq, 1024, 256, 256,
                                       nil, nil, nil, nil, nil);
  // attention
  attn_kernel<<<dim3(64, 8, 8), blk, 0, stream>>>(qp, kp, vt, ctx, colsum);
  // out_seq = ctx @ out_proj_w^T + b
  gemm_bf16<0><<<gg, blk, 0, stream>>>(ctx, sSeq, w_op, opb, 1.0f,
                                       outsq, sSeq, 1024, 256, 256,
                                       nil, nil, nil, nil, nil);
  // y = tgt + silu(BN(out_seq @ fuse_w^T)), stored [b][c][hw]
  gemm_bf16<2><<<gg, blk, 0, stream>>>(outsq, sSeq, w_fu, nil, 1.0f,
                                       out, 256L * 1024, 1024, 256, 256,
                                       tgt, bng, bnb, bnm, bnvr);
  // attn_map
  attnmap_kernel<<<dim3(32), blk, 0, stream>>>(colsum, out + 8L * 256 * 1024, 8192);
}
